// SpMiddleFHD_90486370992724
// MI455X (gfx1250) — matrix operand order repaired and checked by reading, not compiled, not run on hardware
//
#include <hip/hip_runtime.h>
#include <cstddef>

// ---------------- CDNA5 WMMA types ----------------
typedef __attribute__((ext_vector_type(16))) __bf16 v16bf;
typedef __attribute__((ext_vector_type(8)))  float  v8f;

__device__ __forceinline__ unsigned short f2bf(float f) {
  unsigned int u = __float_as_uint(f);
  u += 0x7FFFu + ((u >> 16) & 1u);          // round-to-nearest-even
  return (unsigned short)(u >> 16);
}
__device__ __forceinline__ float bf2f(unsigned short h) {
  return __uint_as_float(((unsigned int)h) << 16);
}

// ---------------- helper kernels ----------------
__global__ void fill_u32_kernel(unsigned int* p, unsigned int v, long long n) {
  long long i = (long long)blockIdx.x * blockDim.x + threadIdx.x;
  if (i < n) p[i] = v;
}

__global__ void scatter_kernel(const float* __restrict__ vf, const int* __restrict__ coords,
                               unsigned short* __restrict__ dense0, float* __restrict__ mask0,
                               int N, int Dd, int Hh, int Ww) {
  int i = blockIdx.x * blockDim.x + threadIdx.x;
  if (i >= N) return;
  int b = coords[4*i+0], z = coords[4*i+1], y = coords[4*i+2], x = coords[4*i+3];
  size_t vox = (((size_t)b * Dd + z) * Hh + y) * (size_t)Ww + x;
  mask0[vox] = 1.0f;
#pragma unroll
  for (int c = 0; c < 4; ++c) dense0[vox*4 + c] = f2bf(vf[4*i + c]);
}

// OIDHW fp32 -> [Kpad][Cout] bf16, K order = ((dz*kh+dy)*kw+dx)*Cin + ci, zero-padded rows
__global__ void convert_weights_kernel(const float* __restrict__ src, unsigned short* __restrict__ dst,
                                       int Cin, int lcin, int Cout, int lcout,
                                       int kd, int kh, int kw, int Kreal, int Kpad) {
  int idx = blockIdx.x * blockDim.x + threadIdx.x;
  if (idx >= Kpad * Cout) return;
  int o = idx & (Cout - 1);
  int k = idx >> lcout;
  unsigned short v = 0;
  if (k < Kreal) {
    int ci  = k & (Cin - 1);
    int tap = k >> lcin;
    int dx = tap % kw; int t = tap / kw;
    int dy = t % kh;   int dz = t / kh;
    v = f2bf(src[((((size_t)o * Cin + ci) * kd + dz) * kh + dy) * kw + dx]);
  }
  dst[idx] = v;
}

// mask_out = (conv(mask_in, ones) > 0)
__global__ void downmask_kernel(const float* __restrict__ mIn, float* __restrict__ mOut,
                                int kd, int kh, int kw, int strD, int strH, int strW,
                                int padD, int padH, int padW,
                                int Din, int Hin, int Win, int Dout, int Hout, int Wout, int Mtot) {
  int i = blockIdx.x * blockDim.x + threadIdx.x;
  if (i >= Mtot) return;
  int x = i % Wout; int t = i / Wout;
  int y = t % Hout; t /= Hout;
  int z = t % Dout; int b = t / Dout;
  int zb = z * strD - padD, yb = y * strH - padH, xb = x * strW - padW;
  float any = 0.0f;
  for (int dz = 0; dz < kd; ++dz) {
    int zi = zb + dz; if ((unsigned)zi >= (unsigned)Din) continue;
    for (int dy = 0; dy < kh; ++dy) {
      int yi = yb + dy; if ((unsigned)yi >= (unsigned)Hin) continue;
      for (int dx = 0; dx < kw; ++dx) {
        int xi = xb + dx; if ((unsigned)xi >= (unsigned)Win) continue;
        any += mIn[(((size_t)b * Din + zi) * Hin + yi) * (size_t)Win + xi];
      }
    }
  }
  mOut[i] = (any > 0.0f) ? 1.0f : 0.0f;
}

// ---------------- implicit-GEMM conv3d with WMMA bf16 ----------------
// Block = 128 threads (4 waves) computing a 64-voxel x Cout macrotile.
// NT = Cout/16 (compile-time: 1,2,4). Wave w owns n-tile (w % NT) and NT
// m-subtiles starting at (w / NT) * NT => NT accumulators, NT back-to-back
// WMMAs per K-step sharing one B fragment. All stage voxel counts are
// multiples of 64.
#define KPAD_MAX 1728
template <int NT>
__global__ __launch_bounds__(128) void conv3d_wmma_kernel(
    const unsigned short* __restrict__ in,
    const unsigned short* __restrict__ wq,
    const float* __restrict__ bnp,          // [4][Cout] gamma,beta,mean,var
    const float* __restrict__ mout,         // output-resolution mask
    unsigned short* __restrict__ out,
    float* __restrict__ finalOut,           // non-null only for last layer
    int Cin, int lcin, int Kreal, int Kpad,
    int kd, int kh, int kw, int strD, int strH, int strW,
    int padD, int padH, int padW,
    int Din, int Hin, int Win, int Dout, int Hout, int Wout, int Mtot)
{
  constexpr int Cout  = NT * 16;
  constexpr int lcout = (NT == 1) ? 4 : (NT == 2) ? 5 : 6;

  __shared__ unsigned short aT[64][40];   // 64 x 32 bf16, padded rows (80B: conflict-free b128)
  __shared__ unsigned short bT[32][72];   // 32 x Cout(<=64) bf16, padded rows (144B)
  __shared__ int sB[64], sZ[64], sY[64], sX[64];  // per-output-voxel base coords
  __shared__ int sTap[KPAD_MAX];                  // packed ci|dz|dy|dx per K index, -1 = pad

  const int tid  = threadIdx.x;
  const int lane = tid & 31;
  const int wv   = tid >> 5;
  const int hsel = lane >> 4;
  const int ml   = lane & 15;
  const int m0   = blockIdx.x << 6;       // 64 voxels per block

  const int nIdx  = wv % NT;              // this wave's n-tile
  const int mBase = (wv / NT) * NT;       // first of this wave's NT m-subtiles

  // ---- one-time per-block precompute: voxel coords + full tap table ----
  for (int m = tid; m < 64; m += 128) {
    int vox = m0 + m;
    if (vox < Mtot) {
      int x = vox % Wout; int t = vox / Wout;
      int y = t % Hout; t /= Hout;
      int z = t % Dout; int b = t / Dout;
      sB[m] = b;
      sZ[m] = z * strD - padD;
      sY[m] = y * strH - padH;
      sX[m] = x * strW - padW;
    } else {
      sB[m] = -1;
    }
  }
  for (int k = tid; k < Kpad; k += 128) {
    int packed = -1;
    if (k < Kreal) {
      int ci  = k & (Cin - 1);
      int tap = k >> lcin;
      int dx = tap % kw; int t = tap / kw;
      int dy = t % kh;   int dz = t / kh;
      packed = ci | (dz << 8) | (dy << 16) | (dx << 24);
    }
    sTap[k] = packed;
  }

  v8f acc[NT];
#pragma unroll
  for (int j = 0; j < NT; ++j)
#pragma unroll
    for (int i = 0; i < 8; ++i) acc[j][i] = 0.0f;

  __syncthreads();

  for (int k0 = 0; k0 < Kpad; k0 += 32) {
    // ---- stage A tile (im2col gather, channels-last => contiguous ci) ----
    for (int idx = tid; idx < 64 * 32; idx += 128) {
      int mm = idx >> 5, kk = idx & 31;
      unsigned short v = 0;
      int pk = sTap[k0 + kk];
      int b  = sB[mm];
      if (pk >= 0 && b >= 0) {
        int ci = pk & 255;
        int zi = sZ[mm] + ((pk >> 8) & 255);
        int yi = sY[mm] + ((pk >> 16) & 255);
        int xi = sX[mm] + (pk >> 24);
        if ((unsigned)zi < (unsigned)Din && (unsigned)yi < (unsigned)Hin &&
            (unsigned)xi < (unsigned)Win)
          v = in[((((size_t)b * Din + zi) * Hin + yi) * (size_t)Win + xi) * Cin + ci];
      }
      aT[mm][kk] = v;
    }
    // ---- stage B tile (pre-transposed weights, fully coalesced) ----
    for (int idx = tid; idx < (Cout << 5); idx += 128) {
      int nn = idx & (Cout - 1), kk = idx >> lcout;
      bT[kk][nn] = wq[(size_t)(k0 + kk) * Cout + nn];
    }
    if (k0 + 32 < Kpad)
      __builtin_prefetch(&wq[(size_t)(k0 + 32) * Cout], 0, 1);   // global_prefetch_b8
    __syncthreads();

    // ---- fragments per documented gfx1250 VGPR layouts ----
    union Frag { v16bf v; uint4 q[2]; };
    // B 32x16 bf16: N col = n0 + ml; elems 0..7 -> K=hsel*8+0..7, 8..15 -> K=16+hsel*8+0..7 (one per wave, reused NT times)
    Frag ub;
    const int nc = (nIdx << 4) + ml;
    unsigned int* ubw = reinterpret_cast<unsigned int*>(&ub.q[0]);
#pragma unroll
    for (int p = 0; p < 8; ++p) {
      const int s = 2 * p;
      const int kr = (s < 8) ? (hsel * 8 + s) : (16 + hsel * 8 + (s - 8));
      ubw[p] = (unsigned int)bT[kr][nc] | ((unsigned int)bT[kr + 1][nc] << 16);
    }
#pragma unroll
    for (int j = 0; j < NT; ++j) {
      // A 16x32 bf16: M row = subtile*16 + ml; elems 0..7 -> K=hsel*8+0..7, 8..15 -> K=16+hsel*8+0..7
      Frag ua;
      const int ar = ((mBase + j) << 4) + ml;
      ua.q[0] = *reinterpret_cast<const uint4*>(&aT[ar][hsel * 8]);
      ua.q[1] = *reinterpret_cast<const uint4*>(&aT[ar][16 + hsel * 8]);
      acc[j] = __builtin_amdgcn_wmma_f32_16x16x32_bf16(
          false, ua.v, false, ub.v, (short)0, acc[j], false, false);
    }
    __syncthreads();
  }

  // ---- fused BN + ReLU + mask epilogue (C/D layout: N = lane&15, M = v + 8*hsel) ----
  const int n = (nIdx << 4) + ml;
  float g  = bnp[n];
  float be = bnp[Cout + n];
  float mu = bnp[2 * Cout + n];
  float va = bnp[3 * Cout + n];
  float sc  = g * rsqrtf(va + 1e-3f);
  float shf = be - mu * sc;
#pragma unroll
  for (int j = 0; j < NT; ++j) {
#pragma unroll
    for (int v = 0; v < 8; ++v) {
      int vox = m0 + ((mBase + j) << 4) + v + (hsel << 3);
      if (vox >= Mtot) continue;
      float val = fmaf(acc[j][v], sc, shf);
      val = fmaxf(val, 0.0f) * mout[vox];
      if (finalOut) {
        int x = vox % Wout; int t = vox / Wout;
        int y = t % Hout; t /= Hout;
        int z = t % Dout; int b = t / Dout;
        // reshape (b,c,d,h,w) -> (b, c*Dout + d, h, w), fp32
        finalOut[(((size_t)b * Cout * Dout + (size_t)n * Dout + z) * Hout + y) * Wout + x] = val;
      } else {
        out[(size_t)vox * Cout + n] = f2bf(val);
      }
    }
  }
}

// ---------------- host orchestration ----------------
static inline size_t alignUp(size_t v, size_t a) { return (v + a - 1) & ~(a - 1); }
static inline int ilog2i(int v) { int l = 0; while ((1 << l) < v) ++l; return l; }

extern "C" void kernel_launch(void* const* d_in, const int* in_sizes, int n_in,
                              void* d_out, int out_size, void* d_ws, size_t ws_size,
                              hipStream_t stream) {
  (void)n_in; (void)out_size; (void)ws_size;
  const int B = 2, D0 = 41, H0 = 256, W0 = 256;
  const float* vf    = (const float*)d_in[0];
  const int* coords  = (const int*)d_in[29];
  const int N = in_sizes[0] / 4;

  struct LD {
    int cin, cout, kd, kh, kw, sd, sh, sw, pd, ph, pw;
    int din, hin, win, dout, hout, wout, mIn, mOut;
  };
  const LD ls[14] = {
    { 4,16, 3,3,3, 1,1,1, 1,1,1, 41,256,256, 41,256,256, 0,0},
    {16,16, 3,3,3, 1,1,1, 1,1,1, 41,256,256, 41,256,256, 0,0},
    {16,32, 3,3,3, 2,2,2, 1,1,1, 41,256,256, 21,128,128, 0,1},
    {32,32, 3,3,3, 1,1,1, 1,1,1, 21,128,128, 21,128,128, 1,1},
    {32,32, 3,3,3, 1,1,1, 1,1,1, 21,128,128, 21,128,128, 1,1},
    {32,64, 3,3,3, 2,2,2, 1,1,1, 21,128,128, 11, 64, 64, 1,2},
    {64,64, 3,3,3, 1,1,1, 1,1,1, 11, 64, 64, 11, 64, 64, 2,2},
    {64,64, 3,3,3, 1,1,1, 1,1,1, 11, 64, 64, 11, 64, 64, 2,2},
    {64,64, 3,3,3, 1,1,1, 1,1,1, 11, 64, 64, 11, 64, 64, 2,2},
    {64,64, 3,3,3, 2,2,2, 0,1,1, 11, 64, 64,  5, 32, 32, 2,3},
    {64,64, 3,3,3, 1,1,1, 1,1,1,  5, 32, 32,  5, 32, 32, 3,3},
    {64,64, 3,3,3, 1,1,1, 1,1,1,  5, 32, 32,  5, 32, 32, 3,3},
    {64,64, 3,3,3, 1,1,1, 1,1,1,  5, 32, 32,  5, 32, 32, 3,3},
    {64,64, 3,1,1, 2,1,1, 0,0,0,  5, 32, 32,  2, 32, 32, 3,4},
  };
  const int SD[5] = {41, 21, 11, 5, 2};
  const int SH[5] = {256, 128, 64, 32, 32};
  const int SW[5] = {256, 128, 64, 32, 32};

  // ---- carve workspace: two ping-pong activation buffers (sized for stage0, 172 MB each),
  //      5 mask buffers, 14 bf16 weight slabs ----
  char* ws = (char*)d_ws;
  size_t cur = 0;
  const size_t vox0 = (size_t)B * D0 * H0 * W0;                      // 5,373,952
  const size_t actBytes = alignUp(vox0 * 16 * sizeof(unsigned short), 256);
  unsigned short* actA = (unsigned short*)(ws + cur); cur += actBytes;
  unsigned short* actB = (unsigned short*)(ws + cur); cur += actBytes;
  float* masks[5];
  for (int s = 0; s < 5; ++s) {
    size_t mv = (size_t)B * SD[s] * SH[s] * SW[s];
    masks[s] = (float*)(ws + cur);
    cur += alignUp(mv * sizeof(float), 256);
  }
  unsigned short* wqs[14]; int Kreal[14], Kpad[14];
  for (int i = 0; i < 14; ++i) {
    Kreal[i] = ls[i].kd * ls[i].kh * ls[i].kw * ls[i].cin;
    Kpad[i]  = (Kreal[i] + 31) & ~31;
    wqs[i] = (unsigned short*)(ws + cur);
    cur += alignUp((size_t)Kpad[i] * ls[i].cout * sizeof(unsigned short), 256);
  }

  // ---- zero dense0 (lives in actA, Cin=4) and mask0, then scatter voxels ----
  {
    long long nw0 = (long long)(vox0 * 2);  // vox0*4 bf16 = vox0*2 u32 words
    fill_u32_kernel<<<dim3((unsigned)((nw0 + 255) / 256)), dim3(256), 0, stream>>>(
        (unsigned int*)actA, 0u, nw0);
    long long nm0 = (long long)vox0;
    fill_u32_kernel<<<dim3((unsigned)((nm0 + 255) / 256)), dim3(256), 0, stream>>>(
        (unsigned int*)masks[0], 0u, nm0);
  }
  scatter_kernel<<<dim3((N + 255) / 256), dim3(256), 0, stream>>>(
      vf, coords, actA, masks[0], N, D0, H0, W0);

  // ---- pre-transpose all weights to bf16 [Kpad][Cout] ----
  for (int i = 0; i < 14; ++i) {
    const float* wsrc = (const float*)d_in[1 + i];
    int elems = Kpad[i] * ls[i].cout;
    convert_weights_kernel<<<dim3((elems + 255) / 256), dim3(256), 0, stream>>>(
        wsrc, wqs[i], ls[i].cin, ilog2i(ls[i].cin), ls[i].cout, ilog2i(ls[i].cout),
        ls[i].kd, ls[i].kh, ls[i].kw, Kreal[i], Kpad[i]);
  }

  // ---- run the 14 layers (ping-pong actA/actB; layer 0 reads dense0 in actA) ----
  for (int i = 0; i < 14; ++i) {
    const LD& l = ls[i];
    int Mtot = B * l.dout * l.hout * l.wout;
    if (l.mOut != l.mIn) {
      downmask_kernel<<<dim3((Mtot + 255) / 256), dim3(256), 0, stream>>>(
          masks[l.mIn], masks[l.mOut],
          l.kd, l.kh, l.kw, l.sd, l.sh, l.sw, l.pd, l.ph, l.pw,
          l.din, l.hin, l.win, l.dout, l.hout, l.wout, Mtot);
    }
    const unsigned short* src = (i % 2 == 0) ? actA : actB;
    unsigned short* dst       = (i % 2 == 0) ? actB : actA;
    float* fin = (i == 13) ? (float*)d_out : nullptr;
    const float* bnp = (const float*)d_in[15 + i];
    dim3 grid((unsigned)((Mtot + 63) / 64)), block(128);
    const int NT = l.cout >> 4;
    switch (NT) {
      case 1:
        conv3d_wmma_kernel<1><<<grid, block, 0, stream>>>(
            src, wqs[i], bnp, masks[l.mOut], dst, fin,
            l.cin, ilog2i(l.cin), Kreal[i], Kpad[i],
            l.kd, l.kh, l.kw, l.sd, l.sh, l.sw, l.pd, l.ph, l.pw,
            l.din, l.hin, l.win, l.dout, l.hout, l.wout, Mtot);
        break;
      case 2:
        conv3d_wmma_kernel<2><<<grid, block, 0, stream>>>(
            src, wqs[i], bnp, masks[l.mOut], dst, fin,
            l.cin, ilog2i(l.cin), Kreal[i], Kpad[i],
            l.kd, l.kh, l.kw, l.sd, l.sh, l.sw, l.pd, l.ph, l.pw,
            l.din, l.hin, l.win, l.dout, l.hout, l.wout, Mtot);
        break;
      default:
        conv3d_wmma_kernel<4><<<grid, block, 0, stream>>>(
            src, wqs[i], bnp, masks[l.mOut], dst, fin,
            l.cin, ilog2i(l.cin), Kreal[i], Kpad[i],
            l.kd, l.kh, l.kw, l.sd, l.sh, l.sw, l.pd, l.ph, l.pw,
            l.din, l.hin, l.win, l.dout, l.hout, l.wout, Mtot);
        break;
    }
  }
}
